// ProFITi_30090540876052
// MI455X (gfx1250) — matrix operand order repaired and checked by reading, not compiled, not run on hardware
//
#include <hip/hip_runtime.h>

typedef __attribute__((ext_vector_type(16))) __bf16        v16bf;
typedef __attribute__((ext_vector_type(8)))  float         v8f;
typedef __attribute__((ext_vector_type(8)))  unsigned int  v8u;

#define B_  64
#define S_  512
#define D_  256
#define F_  4
#define BS_ (B_ * S_)

__device__ __forceinline__ unsigned short f32_to_bf16(float x) {
    unsigned int u = __builtin_bit_cast(unsigned int, x);
    unsigned int r = u + 0x7FFFu + ((u >> 16) & 1u);
    return (unsigned short)(r >> 16);
}
__device__ __forceinline__ float bf16_to_f32(unsigned short h) {
    unsigned int u = ((unsigned int)h) << 16;
    return __builtin_bit_cast(float, u);
}

// ---------------------------------------------------------------------------
// f32 -> bf16 bulk convert (h)
// ---------------------------------------------------------------------------
__global__ void __launch_bounds__(256)
cvt_bf16(const float* __restrict__ in, unsigned short* __restrict__ out, int n) {
    int i = blockIdx.x * 256 + threadIdx.x;
    if (i < n) out[i] = f32_to_bf16(in[i]);
}

// f32 [F,D,D] -> bf16 transposed per layer: out[l][n][k] = in[l][k][n]
__global__ void __launch_bounds__(256)
cvt_w_t(const float* __restrict__ in, unsigned short* __restrict__ out) {
    int i = blockIdx.x * 256 + threadIdx.x;           // F*D*D threads
    int l = i >> 16;
    int r = i & 65535;
    int n = r >> 8;
    int k = r & 255;
    out[(size_t)l * 65536 + (size_t)n * 256 + k] =
        f32_to_bf16(in[(size_t)l * 65536 + (size_t)k * 256 + n]);
}

// ---------------------------------------------------------------------------
// GEMM: C_bf16[M,256] = act(A_bf16[M,256] @ Wt_bf16[N,K]^T + bias[N])
// one wave == 16 rows x 64 cols (4 accumulators); each A fragment feeds
// 4 x v_wmma_f32_16x16x32_bf16, K=256 -> 32 WMMA per wave.
// ---------------------------------------------------------------------------
__global__ void __launch_bounds__(256)
gemm_bf16(const unsigned short* __restrict__ A,
          const unsigned short* __restrict__ Wt,
          const float* __restrict__ bias,
          unsigned short* __restrict__ C,
          int act) {
    const int lane = threadIdx.x;
    const int wave = threadIdx.y;
    const int half = lane >> 4;
    const int ln   = lane & 15;
    const int m0 = (blockIdx.x * 8 + wave) * 16;
    const int n0 = blockIdx.y * 64;

    v8f c[4];
    #pragma unroll
    for (int t = 0; t < 4; ++t) {
        const float bv = bias[n0 + t * 16 + ln];
        #pragma unroll
        for (int v = 0; v < 8; ++v) c[t][v] = bv;
    }
    const unsigned short* Arow = A + (size_t)(m0 + ln) * 256;
    const unsigned short* Wrow[4];
    #pragma unroll
    for (int t = 0; t < 4; ++t)
        Wrow[t] = Wt + (size_t)(n0 + t * 16 + ln) * 256;

    #pragma unroll
    for (int kk = 0; kk < 256; kk += 32) {
        v8u au;
        #pragma unroll
        for (int j = 0; j < 8; ++j) {
            const int ka = ((j >> 2) << 4) + (half << 3) + ((j & 3) << 1);
            au[j] = *(const unsigned int*)(Arow + kk + ka);
        }
        #pragma unroll
        for (int t = 0; t < 4; ++t) {
            v8u bu;
            #pragma unroll
            for (int j = 0; j < 8; ++j)
                bu[j] = *(const unsigned int*)(Wrow[t] + kk + ((j >> 2) << 4) + (half << 3) + ((j & 3) << 1));
            c[t] = __builtin_amdgcn_wmma_f32_16x16x32_bf16(
                       false, __builtin_bit_cast(v16bf, au),
                       false, __builtin_bit_cast(v16bf, bu),
                       (short)0, c[t], false, false);
        }
    }
    #pragma unroll
    for (int t = 0; t < 4; ++t) {
        #pragma unroll
        for (int v = 0; v < 8; ++v) {
            float x = c[t][v];
            if (act) x = fmaxf(x, 0.f);
            const int m = m0 + v + half * 8;
            C[(size_t)m * 256 + n0 + t * 16 + ln] = f32_to_bf16(x);
        }
    }
}

// ---------------------------------------------------------------------------
// Fused triangular attention + matvec + diag ldj. One wave per (batch, 16-row
// tile). Q fragments for the whole row strip are kept resident in VGPRs.
// ---------------------------------------------------------------------------
__global__ void __launch_bounds__(256)
tri_matvec(const unsigned short* __restrict__ Qb,
           const unsigned short* __restrict__ Kb,
           const float* __restrict__ zin,
           const float* __restrict__ mq,
           float* __restrict__ zout,
           float* __restrict__ diag) {
    const int lane = threadIdx.x;
    const int wave = threadIdx.y;
    const int half = lane >> 4;
    const int ln   = lane & 15;
    const int id = blockIdx.x * 8 + wave;
    const int b  = id >> 5;
    const int si = id & 31;
    const int s0 = si * 16;
    const unsigned short* Q = Qb + (size_t)b * S_ * 256;
    const unsigned short* K = Kb + (size_t)b * S_ * 256;
    const float* zb = zin + b * S_;
    const float scale = 0.0625f;  // 256^-0.5

    v8u afr[8];
    {
        const unsigned short* qrow = Q + (size_t)(s0 + ln) * 256;
        #pragma unroll
        for (int kk8 = 0; kk8 < 8; ++kk8) {
            #pragma unroll
            for (int j = 0; j < 8; ++j) {
                const int ka = ((j >> 2) << 4) + (half << 3) + ((j & 3) << 1);
                afr[kk8][j] = *(const unsigned int*)(qrow + kk8 * 32 + ka);
            }
        }
    }

    float racc[8];
    #pragma unroll
    for (int v = 0; v < 8; ++v) racc[v] = 0.f;
    float dlog = 0.f;

    for (int tj = 0; tj <= si; ++tj) {
        const int t0 = tj * 16;
        const unsigned short* krow = K + (size_t)(t0 + ln) * 256;
        if (tj < si)  // uniform branch: prefetch next K tile
            __builtin_prefetch(krow + 16 * 256, 0, 3);

        v8f c;
        #pragma unroll
        for (int v = 0; v < 8; ++v) c[v] = 0.f;

        #pragma unroll
        for (int kk8 = 0; kk8 < 8; ++kk8) {
            v8u bu;
            #pragma unroll
            for (int j = 0; j < 8; ++j)
                bu[j] = *(const unsigned int*)(krow + kk8 * 32 + ((j >> 2) << 4) + (half << 3) + ((j & 3) << 1));
            c = __builtin_amdgcn_wmma_f32_16x16x32_bf16(
                    false, __builtin_bit_cast(v16bf, afr[kk8]),
                    false, __builtin_bit_cast(v16bf, bu),
                    (short)0, c, false, false);
        }

        const float zt = zb[t0 + ln];
        if (tj < si) {
            #pragma unroll
            for (int v = 0; v < 8; ++v) racc[v] += c[v] * scale * zt;
        } else {  // diagonal tile: tril + softplus diagonal + log-det
            #pragma unroll
            for (int v = 0; v < 8; ++v) {
                const int m = v + half * 8;
                const float sc = c[v] * scale;
                if (ln < m) {
                    racc[v] += sc * zt;
                } else if (ln == m) {
                    const float d = (sc > 20.f ? sc : log1pf(__expf(sc))) + 1e-3f;
                    racc[v] += d * zt;
                    dlog += logf(d);
                }
            }
        }
    }

    // reduce over the 16 lanes of each half (rows differ per half)
    #pragma unroll
    for (int v = 0; v < 8; ++v) {
        #pragma unroll
        for (int mask = 1; mask <= 8; mask <<= 1)
            racc[v] += __shfl_xor(racc[v], mask, 32);
    }
    #pragma unroll
    for (int mask = 1; mask <= 16; mask <<= 1)
        dlog += __shfl_xor(dlog, mask, 32);

    if (ln == 0) {
        #pragma unroll
        for (int v = 0; v < 8; ++v) {
            const int s = s0 + v + half * 8;
            zout[b * S_ + s] = racc[v] * mq[b * S_ + s];
        }
        if (half == 0) diag[b * 32 + si] = dlog;
    }
}

// ---------------------------------------------------------------------------
// Head: out[row] = (tanh)(dot(X_bf16[row,:], w2) + b2), one wave per row
// ---------------------------------------------------------------------------
__global__ void __launch_bounds__(256)
head_reduce(const unsigned short* __restrict__ X,
            const float* __restrict__ w2,
            const float* __restrict__ b2,
            float* __restrict__ out,
            int do_tanh) {
    const int lane = threadIdx.x;
    const int row  = blockIdx.x * 8 + threadIdx.y;
    const unsigned short* xr = X + (size_t)row * 256 + lane * 8;
    float acc = 0.f;
    #pragma unroll
    for (int e = 0; e < 8; ++e) acc += bf16_to_f32(xr[e]) * w2[lane * 8 + e];
    #pragma unroll
    for (int mask = 1; mask <= 16; mask <<= 1)
        acc += __shfl_xor(acc, mask, 32);
    if (lane == 0) {
        const float r = acc + b2[0];
        out[row] = do_tanh ? tanhf(r) : r;
    }
}

// ---------------------------------------------------------------------------
// init: el0 = h @ phi_init_w + b ; z0 = (y*mq - el0)*mq. One wave per row.
// ---------------------------------------------------------------------------
__global__ void __launch_bounds__(256)
init_z(const float* __restrict__ h, const float* __restrict__ w,
       const float* __restrict__ bptr, const float* __restrict__ y,
       const float* __restrict__ mq, float* __restrict__ zA) {
    const int lane = threadIdx.x;
    const int row  = blockIdx.x * 8 + threadIdx.y;
    const float* hr = h + (size_t)row * 256 + lane * 8;
    float acc = 0.f;
    #pragma unroll
    for (int e = 0; e < 8; ++e) acc += hr[e] * w[lane * 8 + e];
    #pragma unroll
    for (int mask = 1; mask <= 16; mask <<= 1)
        acc += __shfl_xor(acc, mask, 32);
    if (lane == 0) {
        const float el0 = acc + bptr[0];
        const float m = mq[row];
        zA[row] = (y[row] * m - el0) * m;
    }
}

// ---------------------------------------------------------------------------
// pointwise flow step + deterministic per-batch ldj reduction (block/batch)
// ---------------------------------------------------------------------------
__global__ void __launch_bounds__(256)
pointwise(const float* __restrict__ zB, const float* __restrict__ theta,
          const float* __restrict__ phi, const float* __restrict__ mq,
          const float* __restrict__ diag, float* __restrict__ zA,
          float* __restrict__ ldj, int first) {
    const int b = blockIdx.x;
    const int t = threadIdx.x;
    __shared__ float sm[256];
    float acc = 0.f;
    #pragma unroll
    for (int r = 0; r < 2; ++r) {
        const int idx = b * S_ + t + r * 256;
        const float m  = mq[idx];
        const float th = theta[idx];
        const float z2 = (zB[idx] * __expf(th) + phi[idx]) * m;
        const float ax = fabsf(z2);
        const float lc = ax + log1pf(__expf(-2.f * ax)) - 0.69314718056f; // log cosh
        zA[idx] = sinhf(z2);
        acc += (th + lc) * m;
    }
    sm[t] = acc;
    __syncthreads();
    for (int o = 128; o > 0; o >>= 1) {
        if (t < o) sm[t] += sm[t + o];
        __syncthreads();
    }
    if (t == 0) {
        float d = 0.f;
        #pragma unroll
        for (int j = 0; j < 32; ++j) d += diag[b * 32 + j];
        const float base = first ? 0.f : ldj[b];
        ldj[b] = base + sm[0] + d;
    }
}

__global__ void __launch_bounds__(256)
finalize(const float* __restrict__ z, const float* __restrict__ ldj,
         float* __restrict__ out) {
    const int i = blockIdx.x * 256 + threadIdx.x;
    if (i < BS_) out[i] = z[i];
    else if (i < BS_ + B_) out[i] = ldj[i - BS_];
}

// ---------------------------------------------------------------------------
extern "C" void kernel_launch(void* const* d_in, const int* in_sizes, int n_in,
                              void* d_out, int out_size, void* d_ws, size_t ws_size,
                              hipStream_t stream) {
    (void)in_sizes; (void)n_in; (void)out_size; (void)ws_size;
    const float* h          = (const float*)d_in[0];
    const float* mq         = (const float*)d_in[1];
    const float* y          = (const float*)d_in[2];
    const float* phi_init_w = (const float*)d_in[3];
    const float* phi_init_b = (const float*)d_in[4];
    const float* q_w        = (const float*)d_in[5];
    const float* q_b        = (const float*)d_in[6];
    const float* k_w        = (const float*)d_in[7];
    const float* k_b        = (const float*)d_in[8];
    const float* th_w1      = (const float*)d_in[9];
    const float* th_b1      = (const float*)d_in[10];
    const float* th_w2      = (const float*)d_in[11];
    const float* th_b2      = (const float*)d_in[12];
    const float* ph_w1      = (const float*)d_in[13];
    const float* ph_b1      = (const float*)d_in[14];
    const float* ph_w2      = (const float*)d_in[15];
    const float* ph_b2      = (const float*)d_in[16];
    float* out = (float*)d_out;

    char* wp = (char*)d_ws;
    auto alloc = [&](size_t bytes) -> void* {
        void* p = (void*)wp;
        wp += (bytes + 255) & ~(size_t)255;
        return p;
    };
    const size_t DD = (size_t)D_ * D_;
    unsigned short* hbf = (unsigned short*)alloc((size_t)BS_ * D_ * 2);
    unsigned short* wq  = (unsigned short*)alloc((size_t)F_ * DD * 2);
    unsigned short* wk  = (unsigned short*)alloc((size_t)F_ * DD * 2);
    unsigned short* wt1 = (unsigned short*)alloc((size_t)F_ * DD * 2);
    unsigned short* wp1 = (unsigned short*)alloc((size_t)F_ * DD * 2);
    unsigned short* Qb  = (unsigned short*)alloc((size_t)BS_ * D_ * 2);
    unsigned short* Kb  = (unsigned short*)alloc((size_t)BS_ * D_ * 2);
    unsigned short* T1  = (unsigned short*)alloc((size_t)BS_ * D_ * 2);
    unsigned short* P1  = (unsigned short*)alloc((size_t)BS_ * D_ * 2);
    float* theta = (float*)alloc((size_t)BS_ * 4);
    float* phi   = (float*)alloc((size_t)BS_ * 4);
    float* zA    = (float*)alloc((size_t)BS_ * 4);
    float* zB    = (float*)alloc((size_t)BS_ * 4);
    float* diag  = (float*)alloc((size_t)B_ * 32 * 4);
    float* ldj   = (float*)alloc((size_t)B_ * 4);

    const dim3 wblk(32, 8);

    cvt_bf16<<<(BS_ * D_) / 256, 256, 0, stream>>>(h, hbf, BS_ * D_);
    cvt_w_t<<<(F_ * (int)DD) / 256, 256, 0, stream>>>(q_w, wq);
    cvt_w_t<<<(F_ * (int)DD) / 256, 256, 0, stream>>>(k_w, wk);
    cvt_w_t<<<(F_ * (int)DD) / 256, 256, 0, stream>>>(th_w1, wt1);
    cvt_w_t<<<(F_ * (int)DD) / 256, 256, 0, stream>>>(ph_w1, wp1);
    init_z<<<BS_ / 8, wblk, 0, stream>>>(h, phi_init_w, phi_init_b, y, mq, zA);

    for (int i = 0; i < F_; ++i) {
        const dim3 ggrid(BS_ / 128, D_ / 64);
        gemm_bf16<<<ggrid, wblk, 0, stream>>>(hbf, wq  + i * DD, q_b   + i * D_, Qb, 0);
        gemm_bf16<<<ggrid, wblk, 0, stream>>>(hbf, wk  + i * DD, k_b   + i * D_, Kb, 0);
        gemm_bf16<<<ggrid, wblk, 0, stream>>>(hbf, wt1 + i * DD, th_b1 + i * D_, T1, 1);
        gemm_bf16<<<ggrid, wblk, 0, stream>>>(hbf, wp1 + i * DD, ph_b1 + i * D_, P1, 1);

        head_reduce<<<BS_ / 8, wblk, 0, stream>>>(T1, th_w2 + i * D_, th_b2 + i, theta, 1);
        head_reduce<<<BS_ / 8, wblk, 0, stream>>>(P1, ph_w2 + i * D_, ph_b2 + i, phi, 0);

        tri_matvec<<<(B_ * 32) / 8, wblk, 0, stream>>>(Qb, Kb, zA, mq, zB, diag);
        pointwise<<<B_, 256, 0, stream>>>(zB, theta, phi, mq, diag, zA, ldj, i == 0);
    }
    finalize<<<(BS_ + B_ + 255) / 256, 256, 0, stream>>>(zA, ldj, out);
}
